// MambaDecoder_16673063043258
// MI455X (gfx1250) — matrix operand order repaired and checked by reading, not compiled, not run on hardware
//
#include <hip/hip_runtime.h>

// ---------------------------------------------------------------------------
// MI455X (gfx1250) Mamba decoder. All GEMMs via v_wmma_f32_16x16x32_bf16
// (wave32, fp32 accumulate), 32x32 output tile per wave (2x2 WMMA tiles,
// A/B fragment reuse). Bounds guards are hoisted out of the k-loop so the
// steady state is pure global_load_b128 + v_wmma. Selective scan: 1 thread
// per (b, d_inner) channel, 16 states in VGPRs, 505 sequential steps.
// ---------------------------------------------------------------------------

typedef __bf16 bf16;
typedef __attribute__((ext_vector_type(16))) __bf16 v16bf;
typedef __attribute__((ext_vector_type(8)))  __bf16 bf16x8;
typedef __attribute__((ext_vector_type(8)))  float  v8f;

namespace {
constexpr int kB   = 8;     // batch
constexpr int kT   = 2048;  // raw time steps
constexpr int kC   = 256;   // neural dim
constexpr int kNC  = 41;    // n_classes + 1
constexpr int kDM  = 512;   // d_model
constexpr int kDS  = 16;    // d_state
constexpr int kDC  = 4;     // d_conv
constexpr int kDI  = 1024;  // d_inner
constexpr int kLay = 4;     // layers
constexpr int kDR  = 32;    // dt_rank
constexpr int kL   = 505;   // mamba sequence length  (2048-32)/4 + 1
constexpr int kLP  = 512;   // padded L for xsT row stride
constexpr int kKL  = 32;    // KLEN
constexpr int kStr = 4;     // STRIDE
constexpr int kG   = 20;    // gaussian kernel size
}

// ------------------------------- helpers -----------------------------------
__device__ __forceinline__ float siluf(float x)     { return x / (1.0f + __expf(-x)); }
__device__ __forceinline__ float softplusf(float x) { return (x > 20.0f) ? x : log1pf(__expf(x)); }
__device__ __forceinline__ float squashf(float x)   { return x / (1.0f + fabsf(x)); }

__device__ __forceinline__ v16bf zero_frag() {
  v16bf a;
#pragma unroll
  for (int i = 0; i < 16; ++i) a[i] = (bf16)0.0f;
  return a;
}

__device__ __forceinline__ v16bf pack_frag(bf16x8 lo, bf16x8 hi) {
  v16bf a;
#pragma unroll
  for (int i = 0; i < 8; ++i) { a[i] = lo[i]; a[8 + i] = hi[i]; }
  return a;
}

// A fragment, 16x32 bf16 (ISA 7.12.2): lane m = lane&15, half = lane>>4.
// element e: K = k0 + (e>>3)*16 + half*8 + (e&7)  -> two contiguous 8-elem runs.
__device__ __forceinline__ v16bf load_a_nc(const bf16* __restrict__ A, int lda,
                                           int row0, int k0, int lane) {
  const bf16* p = A + (size_t)(row0 + (lane & 15)) * lda + k0 + (lane >> 4) * 8;
  return pack_frag(*reinterpret_cast<const bf16x8*>(p),
                   *reinterpret_cast<const bf16x8*>(p + 16));
}
__device__ __forceinline__ v16bf load_a_g(const bf16* __restrict__ A, int lda,
                                          int row0, int k0, int lane, int mvalid) {
  int row = row0 + (lane & 15);
  if (row >= mvalid) return zero_frag();
  const bf16* p = A + (size_t)row * lda + k0 + (lane >> 4) * 8;
  return pack_frag(*reinterpret_cast<const bf16x8*>(p),
                   *reinterpret_cast<const bf16x8*>(p + 16));
}

// B fragment, 32x16 bf16: lane n = lane&15, half = lane>>4.
// element e: K = k0 + (e>>3)*16 + half*8 + (e&7). B is W.T with W (N,K) row-major ->
// two contiguous 8-elem runs from W's row n.
__device__ __forceinline__ v16bf load_b_nc(const bf16* __restrict__ W, int ldw,
                                           int col0, int k0, int lane) {
  const bf16* p = W + (size_t)(col0 + (lane & 15)) * ldw + k0 + (lane >> 4) * 8;
  return pack_frag(*reinterpret_cast<const bf16x8*>(p),
                   *reinterpret_cast<const bf16x8*>(p + 16));
}
__device__ __forceinline__ v16bf load_b_g(const bf16* __restrict__ W, int ldw,
                                          int col0, int k0, int lane, int nvalid) {
  int col = col0 + (lane & 15);
  if (col >= nvalid) return zero_frag();
  const bf16* p = W + (size_t)col * ldw + k0 + (lane >> 4) * 8;
  return pack_frag(*reinterpret_cast<const bf16x8*>(p),
                   *reinterpret_cast<const bf16x8*>(p + 16));
}

// Windowed A fragment for the W_in GEMM: logical A[l][f], f = c*32 + j,
// value = xdayT[b][c][l*4 + j]; k0 is a multiple of 32 so both 8-element runs
// are contiguous in t at channel c = k0>>5.
__device__ __forceinline__ v16bf load_w_nc(const bf16* __restrict__ XT,
                                           int row0, int k0, int lane) {
  const bf16* base = XT + (size_t)(k0 >> 5) * kT + (row0 + (lane & 15)) * kStr
                     + (lane >> 4) * 8;
  return pack_frag(*reinterpret_cast<const bf16x8*>(base),
                   *reinterpret_cast<const bf16x8*>(base + 16));
}
__device__ __forceinline__ v16bf load_w_g(const bf16* __restrict__ XT,
                                          int row0, int k0, int lane, int mvalid) {
  int l = row0 + (lane & 15);
  if (l >= mvalid) return zero_frag();
  const bf16* base = XT + (size_t)(k0 >> 5) * kT + l * kStr + (lane >> 4) * 8;
  return pack_frag(*reinterpret_cast<const bf16x8*>(base),
                   *reinterpret_cast<const bf16x8*>(base + 16));
}

// ------------------------------ GEMM kernels --------------------------------
enum { EPI_DAY = 0, EPI_SPLIT = 1, EPI_DBC = 2, EPI_DT = 3, EPI_H = 4, EPI_OUT = 5 };

template <int EPI>
__device__ __forceinline__ void store_tile(v8f acc, int b, int m0, int n0, int lane,
                                           const float* __restrict__ bias,
                                           long long strideBias,
                                           void* __restrict__ C0, long long sC0,
                                           void* __restrict__ C1, long long sC1,
                                           int M, int N) {
  int n    = n0 + (lane & 15);
  int half = lane >> 4;
  float bv = 0.0f;
  if (bias != nullptr && n < N) bv = bias[(size_t)b * strideBias + n];
#pragma unroll
  for (int r = 0; r < 8; ++r) {
    int m = m0 + r + (half << 3);
    if (m >= M || n >= N) continue;
    float v = acc[r] + bv;
    if (EPI == EPI_DAY) {            // +bias, squash, store bf16 transposed [c][t]
      ((bf16*)C0)[(size_t)b * sC0 + (size_t)n * kT + m] = (bf16)squashf(v);
    } else if (EPI == EPI_SPLIT) {   // in_proj: xs -> transposed [d][l]; z -> silu
      if (n < kDI)
        ((bf16*)C0)[(size_t)b * sC0 + (size_t)n * kLP + m] = (bf16)v;
      else
        ((bf16*)C1)[(size_t)b * sC1 + (size_t)m * kDI + (n - kDI)] = (bf16)siluf(v);
    } else if (EPI == EPI_DBC) {     // x_proj: fp32 dbc + bf16 copy of dt columns
      ((float*)C0)[(size_t)b * sC0 + (size_t)m * 64 + n] = v;
      if (n < kDR)
        ((bf16*)C1)[(size_t)b * sC1 + (size_t)m * kDR + n] = (bf16)v;
    } else if (EPI == EPI_DT) {      // dt_proj: +bias, softplus, fp32
      ((float*)C0)[(size_t)b * sC0 + (size_t)m * kDI + n] = softplusf(v);
    } else if (EPI == EPI_H) {       // plain bf16 row-major
      ((bf16*)C0)[(size_t)b * sC0 + (size_t)m * N + n] = (bf16)v;
    } else {                         // EPI_OUT: +bias, fp32 row-major (head)
      ((float*)C0)[(size_t)b * sC0 + (size_t)m * N + n] = v;
    }
  }
}

#define WMMA_BF16(A_, B_, C_) \
  __builtin_amdgcn_wmma_f32_16x16x32_bf16(false, (A_), false, (B_), (short)0, (C_), false, false)

// C[M,N] = epilogue(A[M,K] @ W[N,K]^T + bias). One wave = one 32x32 tile
// (2x2 WMMA sub-tiles, A and B fragments each reused twice).
template <int EPI>
__global__ __launch_bounds__(256) void gemm_std(
    const bf16* __restrict__ A, int lda, long long strideA,
    const bf16* __restrict__ W, int ldw, long long strideW,
    const float* __restrict__ bias, long long strideBias,
    void* __restrict__ C0, long long strideC0,
    void* __restrict__ C1, long long strideC1,
    int M, int N, int K, int tilesN) {
  int b    = blockIdx.y;
  int wave = (blockIdx.x * blockDim.x + threadIdx.x) >> 5;
  int lane = threadIdx.x & 31;
  int tiles = ((M + 31) >> 5) * tilesN;
  if (wave >= tiles) return;           // wave-uniform exit
  int m0 = (wave / tilesN) << 5;
  int n0 = (wave % tilesN) << 5;

  const bf16* Ab = A + (size_t)b * strideA;
  const bf16* Wb = W + (size_t)b * strideW;

  v8f acc00, acc01, acc10, acc11;
#pragma unroll
  for (int r = 0; r < 8; ++r) { acc00[r] = 0.f; acc01[r] = 0.f; acc10[r] = 0.f; acc11[r] = 0.f; }

  if (m0 + 32 <= M && n0 + 32 <= N) {
    // Fast path: no guards, EXEC all-1s, pure load+wmma steady state.
    for (int k0 = 0; k0 < K; k0 += 32) {
      v16bf a0 = load_a_nc(Ab, lda, m0,      k0, lane);
      v16bf a1 = load_a_nc(Ab, lda, m0 + 16, k0, lane);
      v16bf b0 = load_b_nc(Wb, ldw, n0,      k0, lane);
      v16bf b1 = load_b_nc(Wb, ldw, n0 + 16, k0, lane);
      acc00 = WMMA_BF16(a0, b0, acc00);
      acc01 = WMMA_BF16(a0, b1, acc01);
      acc10 = WMMA_BF16(a1, b0, acc10);
      acc11 = WMMA_BF16(a1, b1, acc11);
    }
  } else {
    for (int k0 = 0; k0 < K; k0 += 32) {
      v16bf a0 = load_a_g(Ab, lda, m0,      k0, lane, M);
      v16bf a1 = load_a_g(Ab, lda, m0 + 16, k0, lane, M);
      v16bf b0 = load_b_g(Wb, ldw, n0,      k0, lane, N);
      v16bf b1 = load_b_g(Wb, ldw, n0 + 16, k0, lane, N);
      acc00 = WMMA_BF16(a0, b0, acc00);
      acc01 = WMMA_BF16(a0, b1, acc01);
      acc10 = WMMA_BF16(a1, b0, acc10);
      acc11 = WMMA_BF16(a1, b1, acc11);
    }
  }

  store_tile<EPI>(acc00, b, m0,      n0,      lane, bias, strideBias, C0, strideC0, C1, strideC1, M, N);
  store_tile<EPI>(acc01, b, m0,      n0 + 16, lane, bias, strideBias, C0, strideC0, C1, strideC1, M, N);
  store_tile<EPI>(acc10, b, m0 + 16, n0,      lane, bias, strideBias, C0, strideC0, C1, strideC1, M, N);
  store_tile<EPI>(acc11, b, m0 + 16, n0 + 16, lane, bias, strideBias, C0, strideC0, C1, strideC1, M, N);
}

// W_in GEMM with the strided-window A operand (K = 256*32 = 8192, N = 512).
__global__ __launch_bounds__(256) void gemm_win(
    const bf16* __restrict__ XT, long long strideX,
    const bf16* __restrict__ W, int ldw,
    const float* __restrict__ bias,
    bf16* __restrict__ C0, long long strideC0,
    int M, int N, int K, int tilesN) {
  int b    = blockIdx.y;
  int wave = (blockIdx.x * blockDim.x + threadIdx.x) >> 5;
  int lane = threadIdx.x & 31;
  int tiles = ((M + 31) >> 5) * tilesN;
  if (wave >= tiles) return;
  int m0 = (wave / tilesN) << 5;
  int n0 = (wave % tilesN) << 5;

  const bf16* Xb = XT + (size_t)b * strideX;

  v8f acc00, acc01, acc10, acc11;
#pragma unroll
  for (int r = 0; r < 8; ++r) { acc00[r] = 0.f; acc01[r] = 0.f; acc10[r] = 0.f; acc11[r] = 0.f; }

  if (m0 + 32 <= M) {   // N = 512 is always full
    for (int k0 = 0; k0 < K; k0 += 32) {
      v16bf a0 = load_w_nc(Xb, m0,      k0, lane);
      v16bf a1 = load_w_nc(Xb, m0 + 16, k0, lane);
      v16bf b0 = load_b_nc(W, ldw, n0,      k0, lane);
      v16bf b1 = load_b_nc(W, ldw, n0 + 16, k0, lane);
      acc00 = WMMA_BF16(a0, b0, acc00);
      acc01 = WMMA_BF16(a0, b1, acc01);
      acc10 = WMMA_BF16(a1, b0, acc10);
      acc11 = WMMA_BF16(a1, b1, acc11);
    }
  } else {
    for (int k0 = 0; k0 < K; k0 += 32) {
      v16bf a0 = load_w_g(Xb, m0,      k0, lane, M);
      v16bf a1 = load_w_g(Xb, m0 + 16, k0, lane, M);
      v16bf b0 = load_b_nc(W, ldw, n0,      k0, lane);
      v16bf b1 = load_b_nc(W, ldw, n0 + 16, k0, lane);
      acc00 = WMMA_BF16(a0, b0, acc00);
      acc01 = WMMA_BF16(a0, b1, acc01);
      acc10 = WMMA_BF16(a1, b0, acc10);
      acc11 = WMMA_BF16(a1, b1, acc11);
    }
  }

  int n    = lane & 15;
  int half = lane >> 4;
  float bv0 = bias[n0 + n];
  float bv1 = bias[n0 + 16 + n];
#pragma unroll
  for (int r = 0; r < 8; ++r) {
    int m0r = m0 + r + (half << 3);
    int m1r = m0 + 16 + r + (half << 3);
    if (m0r < M) {
      C0[(size_t)b * strideC0 + (size_t)m0r * N + n0 + n]      = (bf16)(acc00[r] + bv0);
      C0[(size_t)b * strideC0 + (size_t)m0r * N + n0 + 16 + n] = (bf16)(acc01[r] + bv1);
    }
    if (m1r < M) {
      C0[(size_t)b * strideC0 + (size_t)m1r * N + n0 + n]      = (bf16)(acc10[r] + bv0);
      C0[(size_t)b * strideC0 + (size_t)m1r * N + n0 + 16 + n] = (bf16)(acc11[r] + bv1);
    }
  }
}

// --------------------------- elementwise kernels ----------------------------
__global__ void cvt_f32_bf16(const float* __restrict__ src, bf16* __restrict__ dst, size_t n) {
  size_t i = (size_t)blockIdx.x * blockDim.x + threadIdx.x;
  size_t s = (size_t)gridDim.x * blockDim.x;
  for (; i < n; i += s) dst[i] = (bf16)src[i];
}

// Transpose the selected day matrix into (N,K) bf16 and gather its bias.
__global__ void day_w_kernel(const float* __restrict__ dayW, const float* __restrict__ dayBias,
                             const int* __restrict__ dayIdx,
                             bf16* __restrict__ wt, float* __restrict__ biasB) {
  int b   = blockIdx.y;
  int day = dayIdx[b];
  int i   = blockIdx.x * blockDim.x + threadIdx.x;  // over 256*256
  if (i >= kC * kC) return;
  int n = i >> 8, k = i & 255;
  wt[(size_t)b * kC * kC + (size_t)n * kC + k] =
      (bf16)dayW[(size_t)day * kC * kC + (size_t)k * kC + n];
  if (k == 0) biasB[b * kC + n] = dayBias[(size_t)day * kC + n];
}

// Gaussian depthwise smoothing (SAME, pad_lo=9), output bf16 [b][t][c].
__global__ void gauss_kernel(const float* __restrict__ x, bf16* __restrict__ out) {
  __shared__ float gk[kG];
  if (threadIdx.x < kG) {
    float m = (float)threadIdx.x - (kG - 1) * 0.5f;
    gk[threadIdx.x] = __expf(-0.5f * (m / 2.0f) * (m / 2.0f));
  }
  __syncthreads();
  float s = 0.0f;
#pragma unroll
  for (int j = 0; j < kG; ++j) s += gk[j];

  size_t idx = (size_t)blockIdx.x * blockDim.x + threadIdx.x;
  if (idx >= (size_t)kB * kT * kC) return;
  int c = (int)(idx % kC);
  int t = (int)((idx / kC) % kT);
  int b = (int)(idx / ((size_t)kC * kT));
  float acc = 0.0f;
#pragma unroll
  for (int j = 0; j < kG; ++j) {
    int tt = t + j - 9;
    float xv = (tt >= 0 && tt < kT) ? x[((size_t)b * kT + tt) * kC + c] : 0.0f;
    acc += xv * gk[j];
  }
  out[idx] = (bf16)(acc / s);
}

// Causal depthwise conv (width 4) + bias + silu: xsT[b][d][l] -> xconv[b][l][d].
__global__ void conv_silu_kernel(const bf16* __restrict__ xsT, const float* __restrict__ cw,
                                 const float* __restrict__ cb, bf16* __restrict__ out) {
  size_t idx = (size_t)blockIdx.x * blockDim.x + threadIdx.x;
  if (idx >= (size_t)kB * kDI * kL) return;
  int l = (int)(idx % kL);
  int d = (int)((idx / kL) % kDI);
  int b = (int)(idx / ((size_t)kL * kDI));
  const bf16* row = xsT + ((size_t)b * kDI + d) * kLP;
  float acc = cb[d];
#pragma unroll
  for (int j = 0; j < kDC; ++j) {
    int ll = l - (kDC - 1) + j;
    if (ll >= 0) acc += (float)row[ll] * cw[d * kDC + j];
  }
  out[((size_t)b * kL + l) * kDI + d] = (bf16)siluf(acc);
}

// Selective scan: 1 thread per (b,d) channel, 16 states in VGPRs.
// y[b,l,d] = (scan_y + u*D) * silu(z), written bf16 for the out_proj GEMM.
__global__ __launch_bounds__(256) void scan_kernel(
    const float* __restrict__ dbc,   // [B][L][64]   (B_t = 32..47, C_t = 48..63)
    const float* __restrict__ dtb,   // [B][L][1024] softplus'ed dt
    const bf16*  __restrict__ u,     // [B][L][1024] post-conv silu'ed xs
    const bf16*  __restrict__ zs,    // [B][L][1024] silu(z)
    const float* __restrict__ A_log, // [1024][16] (layer slice)
    const float* __restrict__ Dp,    // [1024]     (layer slice)
    bf16* __restrict__ y) {          // [B][L][1024]
  int b = blockIdx.x >> 2;
  int d = ((blockIdx.x & 3) << 8) + threadIdx.x;
  __shared__ float sBC[32];

  float Av[kDS], h[kDS];
#pragma unroll
  for (int s = 0; s < kDS; ++s) {
    Av[s] = -__expf(A_log[d * kDS + s]);
    h[s]  = 0.0f;
  }
  float Dv = Dp[d];

  for (int l = 0; l < kL; ++l) {
    size_t row = (size_t)b * kL + l;
    if (threadIdx.x < 32) sBC[threadIdx.x] = dbc[row * 64 + 32 + threadIdx.x];
    __syncthreads();
    float dt = dtb[row * kDI + d];
    float uu = (float)u[row * kDI + d];
    float du = dt * uu;
    float yo = 0.0f;
#pragma unroll
    for (int s = 0; s < kDS; ++s) {
      h[s] = __expf(dt * Av[s]) * h[s] + du * sBC[s];
      yo  += h[s] * sBC[16 + s];
    }
    yo += uu * Dv;
    y[row * kDI + d] = (bf16)(yo * (float)zs[row * kDI + d]);
    __syncthreads();
  }
}

// ------------------------------- launcher -----------------------------------
extern "C" void kernel_launch(void* const* d_in, const int* in_sizes, int n_in,
                              void* d_out, int out_size, void* d_ws, size_t ws_size,
                              hipStream_t stream) {
  // Input order: setup_inputs() top-level, params flattened jax-style (sorted keys).
  const float* neuralInput = (const float*)d_in[0];   // (8,2048,256)
  const int*   dayIdx      = (const int*)d_in[1];     // (8,)
  const float* W_in        = (const float*)d_in[2];   // (512, 8192)
  const float* W_out       = (const float*)d_in[3];   // (41, 512)
  const float* b_in        = (const float*)d_in[4];   // (512,)
  const float* b_out       = (const float*)d_in[5];   // (41,)
  const float* dayBias     = (const float*)d_in[6];   // (24,1,256)
  const float* dayWeights  = (const float*)d_in[7];   // (24,256,256)
  const float* A_log       = (const float*)d_in[8];   // (4,1024,16)
  const float* Dparam      = (const float*)d_in[9];   // (4,1024)
  const float* conv_b      = (const float*)d_in[10];  // (4,1024)
  const float* conv_w      = (const float*)d_in[11];  // (4,1024,4)
  const float* dt_proj_b   = (const float*)d_in[12];  // (4,1024)
  const float* dt_proj_w   = (const float*)d_in[13];  // (4,1024,32)
  const float* in_proj     = (const float*)d_in[14];  // (4,2048,512)
  const float* out_proj    = (const float*)d_in[15];  // (4,512,1024)
  const float* x_proj      = (const float*)d_in[16];  // (4,64,1024)
  float* outp = (float*)d_out;                        // (8,505,41)

  // Workspace carve-up (~95 MB), 256-byte aligned.
  char* wsb = (char*)d_ws;
  size_t off = 0;
  auto alloc = [&](size_t bytes) -> char* {
    char* p = wsb + off;
    off = (off + bytes + 255) & ~(size_t)255;
    return p;
  };
  bf16*  smB     = (bf16*)alloc((size_t)kB * kT * kC * 2);        // smoothed, bf16
  bf16*  xdayT   = (bf16*)alloc((size_t)kB * kC * kT * 2);        // day-transformed, [b][c][t]
  bf16*  wInB    = (bf16*)alloc((size_t)kDM * kC * kKL * 2);      // W_in bf16
  bf16*  wOutB   = (bf16*)alloc((size_t)kNC * kDM * 2);           // W_out bf16
  bf16*  inProjB = (bf16*)alloc((size_t)kLay * 2 * kDI * kDM * 2);
  bf16*  xProjB  = (bf16*)alloc((size_t)kLay * 64 * kDI * 2);
  bf16*  dtWB    = (bf16*)alloc((size_t)kLay * kDI * kDR * 2);
  bf16*  outPB   = (bf16*)alloc((size_t)kLay * kDM * kDI * 2);
  bf16*  dayWtB  = (bf16*)alloc((size_t)kB * kC * kC * 2);        // per-batch W.T bf16
  float* dayBB   = (float*)alloc((size_t)kB * kC * 4);            // per-batch bias
  bf16*  hA      = (bf16*)alloc((size_t)kB * kL * kDM * 2);       // h ping
  bf16*  hBb     = (bf16*)alloc((size_t)kB * kL * kDM * 2);       // h pong
  bf16*  xsT     = (bf16*)alloc((size_t)kB * kDI * kLP * 2);      // xs transposed
  bf16*  zsB     = (bf16*)alloc((size_t)kB * kL * kDI * 2);       // silu(z)
  bf16*  xconv   = (bf16*)alloc((size_t)kB * kL * kDI * 2);       // silu(conv(xs))
  float* dbc     = (float*)alloc((size_t)kB * kL * 64 * 4);       // x_proj output f32
  bf16*  dbc32   = (bf16*)alloc((size_t)kB * kL * kDR * 2);       // dt columns bf16
  float* dtbuf   = (float*)alloc((size_t)kB * kL * kDI * 4);      // softplus(dt)
  bf16*  ybuf    = (bf16*)alloc((size_t)kB * kL * kDI * 2);       // gated scan output
  (void)ws_size; (void)n_in; (void)in_sizes; (void)out_size;

  // 1) Weight conversions (bf16 copies live in L2 for the whole launch).
  cvt_f32_bf16<<<1024, 256, 0, stream>>>(W_in,      wInB,    (size_t)kDM * kC * kKL);
  cvt_f32_bf16<<<64,   256, 0, stream>>>(W_out,     wOutB,   (size_t)kNC * kDM);
  cvt_f32_bf16<<<1024, 256, 0, stream>>>(in_proj,   inProjB, (size_t)kLay * 2 * kDI * kDM);
  cvt_f32_bf16<<<256,  256, 0, stream>>>(x_proj,    xProjB,  (size_t)kLay * 64 * kDI);
  cvt_f32_bf16<<<128,  256, 0, stream>>>(dt_proj_w, dtWB,    (size_t)kLay * kDI * kDR);
  cvt_f32_bf16<<<1024, 256, 0, stream>>>(out_proj,  outPB,   (size_t)kLay * kDM * kDI);
  day_w_kernel<<<dim3((kC * kC + 255) / 256, kB), 256, 0, stream>>>(
      dayWeights, dayBias, dayIdx, dayWtB, dayBB);

  // 2) Gaussian smoothing.
  gauss_kernel<<<(int)(((size_t)kB * kT * kC + 255) / 256), 256, 0, stream>>>(neuralInput, smB);

  auto blocksFor = [](int M, int N) {
    int tiles = ((M + 31) / 32) * ((N + 31) / 32);
    return (tiles + 7) / 8;  // 8 waves (32x32 tiles) per 256-thread block
  };
  auto tilesN32 = [](int N) { return (N + 31) / 32; };

  // 3) Day transform GEMM: [2048,256]@[256,256] per batch, squash, store [c][t].
  gemm_std<EPI_DAY><<<dim3(blocksFor(kT, kC), kB), 256, 0, stream>>>(
      smB, kC, (long long)kT * kC,
      dayWtB, kC, (long long)kC * kC,
      dayBB, kC,
      xdayT, (long long)kC * kT, nullptr, 0,
      kT, kC, kC, tilesN32(kC));

  // 4) Windowed W_in GEMM: [505, 8192]@[8192, 512] per batch + b_in -> h (bf16).
  gemm_win<<<dim3(blocksFor(kL, kDM), kB), 256, 0, stream>>>(
      xdayT, (long long)kC * kT,
      wInB, kC * kKL, b_in,
      hA, (long long)kL * kDM,
      kL, kDM, kC * kKL, tilesN32(kDM));

  // 5) Mamba layers.
  bf16* hcur = hA;
  bf16* hnxt = hBb;
  for (int i = 0; i < kLay; ++i) {
    // 5a) in_proj: [505,512]@[512,2048]; xs -> xsT (transposed), z -> silu -> zsB.
    gemm_std<EPI_SPLIT><<<dim3(blocksFor(kL, 2 * kDI), kB), 256, 0, stream>>>(
        hcur, kDM, (long long)kL * kDM,
        inProjB + (size_t)i * 2 * kDI * kDM, kDM, 0,
        nullptr, 0,
        xsT, (long long)kDI * kLP, zsB, (long long)kL * kDI,
        kL, 2 * kDI, kDM, tilesN32(2 * kDI));

    // 5b) depthwise causal conv + silu.
    conv_silu_kernel<<<(int)(((size_t)kB * kDI * kL + 255) / 256), 256, 0, stream>>>(
        xsT, conv_w + (size_t)i * kDI * kDC, conv_b + (size_t)i * kDI, xconv);

    // 5c) x_proj: [505,1024]@[1024,64] -> dbc (f32) + dt columns (bf16).
    gemm_std<EPI_DBC><<<dim3(blocksFor(kL, 64), kB), 256, 0, stream>>>(
        xconv, kDI, (long long)kL * kDI,
        xProjB + (size_t)i * 64 * kDI, kDI, 0,
        nullptr, 0,
        dbc, (long long)kL * 64, dbc32, (long long)kL * kDR,
        kL, 64, kDI, tilesN32(64));

    // 5d) dt_proj: [505,32]@[32,1024] + bias, softplus -> dtbuf (f32).
    gemm_std<EPI_DT><<<dim3(blocksFor(kL, kDI), kB), 256, 0, stream>>>(
        dbc32, kDR, (long long)kL * kDR,
        dtWB + (size_t)i * kDI * kDR, kDR, 0,
        dt_proj_b + (size_t)i * kDI, 0,
        dtbuf, (long long)kL * kDI, nullptr, 0,
        kL, kDI, kDR, tilesN32(kDI));

    // 5e) selective scan + D skip + silu(z) gating -> ybuf (bf16).
    scan_kernel<<<kB * (kDI / 256), 256, 0, stream>>>(
        dbc, dtbuf, xconv, zsB,
        A_log + (size_t)i * kDI * kDS, Dparam + (size_t)i * kDI, ybuf);

    // 5f) out_proj: [505,1024]@[1024,512] -> next h (bf16).
    gemm_std<EPI_H><<<dim3(blocksFor(kL, kDM), kB), 256, 0, stream>>>(
        ybuf, kDI, (long long)kL * kDI,
        outPB + (size_t)i * kDM * kDI, kDI, 0,
        nullptr, 0,
        hnxt, (long long)kL * kDM, nullptr, 0,
        kL, kDM, kDI, tilesN32(kDM));

    bf16* t = hcur; hcur = hnxt; hnxt = t;
  }

  // 6) Head: [505,512]@[512,41] + b_out -> d_out (f32).
  gemm_std<EPI_OUT><<<dim3(blocksFor(kL, kNC), kB), 256, 0, stream>>>(
      hcur, kDM, (long long)kL * kDM,
      wOutB, kDM, 0,
      b_out, 0,
      outp, (long long)kL * kNC, nullptr, 0,
      kL, kNC, kDM, tilesN32(kNC));
}
